// MyConv2d_19748259627224
// MI455X (gfx1250) — matrix operand order repaired and checked by reading, not compiled, not run on hardware
//
#include <hip/hip_runtime.h>
#include <stdint.h>

// ---------------------------------------------------------------------------
// x(32,256,56,56) * softquant(p_c)(256,256,3,3), stride 1, pad 1
// Implicit GEMM: M = 32*56*56 = 100352, N = 256, K = 256*9 = 2304.
// ---------------------------------------------------------------------------
#define C_IN     256
#define C_OUT    256
#define KWIN     9               // 3x3
#define KDIM     (C_IN * KWIN)   // 2304
#define IMG_W    56
#define IMG_HW   3136            // 56*56
#define BATCH    32
#define M_TOTAL  (BATCH * IMG_HW) // 100352
#define LVL      7
#define BETA     10.0f

#define BLK_M    64
#define BLK_N    256             // whole Co per block -> weights L2-resident
#define KSTEP    32
#define NITER    (KDIM / KSTEP)  // 72
#define LDS_STRIDE 40            // bf16 elems per A-row (80B, 16B aligned)

typedef __attribute__((ext_vector_type(16))) __bf16 v16bf;
typedef __attribute__((ext_vector_type(8)))  __bf16 v8bf;
typedef __attribute__((ext_vector_type(8)))  float  v8f;
typedef __attribute__((ext_vector_type(4)))  unsigned int v4u;
typedef __attribute__((ext_vector_type(8)))  int v8i_t;
typedef __attribute__((ext_vector_type(4)))  int v4i_t;

#if defined(__has_builtin)
#if __has_builtin(__builtin_amdgcn_tensor_load_to_lds) && \
    __has_builtin(__builtin_amdgcn_s_wait_tensorcnt)
#define USE_TDM 1
#endif
#endif
#ifndef USE_TDM
#define USE_TDM 0
#endif

// ---------------------------------------------------------------------------
// Kernel 1: soft-quantize weights -> bf16 B-matrix, layout [Co][Ci*9].
// ---------------------------------------------------------------------------
__global__ __launch_bounds__(256)
void quantize_weights(const float* __restrict__ p_c,
                      const float* __restrict__ q_level,
                      __bf16* __restrict__ wq)
{
    int e = blockIdx.x * 256 + threadIdx.x;        // exact grid: 2304*256
    const float* p = p_c + e * LVL;

    float v[LVL], q[LVL];
    float ss = 0.0f;
#pragma unroll
    for (int l = 0; l < LVL; ++l) { v[l] = p[l]; q[l] = q_level[l]; ss += v[l] * v[l]; }

    float scale = BETA * rsqrtf(ss);               // beta / ||p||
    float mx = v[0] * scale;
#pragma unroll
    for (int l = 1; l < LVL; ++l) { float z = v[l] * scale; mx = z > mx ? z : mx; }

    float sum = 0.0f, acc = 0.0f;
#pragma unroll
    for (int l = 0; l < LVL; ++l) {
        float ez = __expf(v[l] * scale - mx);
        sum += ez;
        acc += ez * q[l];
    }
    wq[e] = (__bf16)(acc / sum);                   // native cvt
}

// ---------------------------------------------------------------------------
// im2col helpers (A tile): thread fills m = tid&63, kk = (tid>>6)*8 + j.
// ---------------------------------------------------------------------------
__device__ __forceinline__ void im2col_load(const float* __restrict__ x,
                                            int x_img, int hrow, int wcol,
                                            int kstart, float (&vals)[8])
{
    int ci = kstart / KWIN;                        // one division per K-step
    int rs = kstart - ci * KWIN;                   // 0..8
#pragma unroll
    for (int j = 0; j < 8; ++j) {
        const int r  = (rs * 11) >> 5;             // rs / 3
        const int s  = rs - 3 * r;
        const int hi = hrow + r - 1;
        const int wi = wcol + s - 1;
        vals[j] = 0.0f;
        if ((unsigned)hi < IMG_W && (unsigned)wi < IMG_W)
            vals[j] = x[x_img + ci * IMG_HW + hi * IMG_W + wi];
        ++rs;
        if (rs == KWIN) { rs = 0; ++ci; }
    }
}

__device__ __forceinline__ void im2col_store(__bf16* __restrict__ lbuf,
                                             int m_loc, int kk_base,
                                             const float (&vals)[8])
{
    union { uint4 u; v8bf v; } p;
#pragma unroll
    for (int j = 0; j < 8; ++j) p.v[j] = (__bf16)vals[j];   // hw cvt (packed)
    *(uint4*)&lbuf[m_loc * LDS_STRIDE + kk_base] = p.u;     // 1x ds_store_b128
}

#if USE_TDM
// ---------------------------------------------------------------------------
// TDM: stream B K-slice (tile 32(k) x 256(n) bf16 = 16 KB) global -> LDS.
// D# per ISA 8.3/8.4: 2-D tensor [n][k] bf16, dim0(k)=2304 stride 2304,
// dim1(n)=256; tile_dim0=32, tile_dim1=256. Groups 2/3 zero (2-D tile).
// Result in LDS is n-major: lB[n*32 + kk].
// ---------------------------------------------------------------------------
__device__ __forceinline__ void tdm_load_B(const __bf16* __restrict__ wq,
                                           int k0, __bf16* ldsB)
{
    const unsigned long long ga = (unsigned long long)(uintptr_t)(wq + k0);
    const unsigned lds = (unsigned)(uintptr_t)ldsB;   // low 32 bits = LDS offset

    v4u g0;
    g0[0] = 1u;                                        // count=1, user D#
    g0[1] = lds;                                       // lds_addr
    g0[2] = (unsigned)(ga & 0xFFFFFFFFu);              // global_addr[31:0]
    g0[3] = (unsigned)((ga >> 32) & 0x01FFFFFFu)       // global_addr[56:32]
          | (2u << 30);                                // type = 2 ("image")

    v8i_t g1;
    g1[0] = (1 << 16);                                 // data_size=1 (2 bytes)
    g1[1] = (int)((KDIM & 0xFFFFu) << 16);             // tensor_dim0 lo16
    g1[2] = (int)(((unsigned)KDIM >> 16)               // tensor_dim0 hi16
          | ((BLK_N & 0xFFFFu) << 16));                // tensor_dim1 lo16
    g1[3] = (int)(((unsigned)BLK_N >> 16)              // tensor_dim1 hi16
          | ((unsigned)KSTEP << 16));                  // tile_dim0 = 32
    g1[4] = BLK_N;                                     // tile_dim1=256, dim2=0
    g1[5] = KDIM;                                      // tensor_dim0_stride lo
    g1[6] = 0;                                         // stride hi | dim1_stride
    g1[7] = 0;

    const v4i_t gz4 = {0, 0, 0, 0};
    const v8i_t gz8 = {0, 0, 0, 0, 0, 0, 0, 0};
    // 6-arg toolchain variant: (v4u, v8i, v4i, v4i, v8i, i32 cpol)
    __builtin_amdgcn_tensor_load_to_lds(g0, g1, gz4, gz4, gz8, 0);
}
#endif

// ---------------------------------------------------------------------------
// Kernel 2: implicit-GEMM conv, bf16 WMMA, double-buffered LDS pipeline.
// 8 waves; wave (wm 0..1, wn 0..3) computes 32(M) x 64(N):
// 2 A-frags x 4 B-frags -> 8 v_wmma_f32_16x16x32_bf16 per K-step.
// A: register-staged im2col -> LDS.  B: TDM tensor_load_to_lds (double buf).
// ---------------------------------------------------------------------------
__global__ __launch_bounds__(256)
void conv_wmma(const float* __restrict__ x,
               const __bf16* __restrict__ wq,
               float* __restrict__ out)
{
    __shared__ __align__(16) __bf16 lA[2][BLK_M * LDS_STRIDE];  // 10 KB
#if USE_TDM
    __shared__ __align__(64) __bf16 lB[2][BLK_N * KSTEP];       // 32 KB
#endif

    const int tid    = threadIdx.x;
    const int lane   = tid & 31;
    const int wave   = tid >> 5;
    const int wm     = wave & 1;           // 0..1 : rows wm*32
    const int wn     = wave >> 1;          // 0..3 : cols wn*64
    const int laneLo = lane & 15;
    const int laneHi = lane >> 4;          // 0 or 1

    const int blockM  = blockIdx.x * BLK_M;          // 1568 blocks
    const int b_img   = blockM / IMG_HW;             // 3136 % 64 == 0
    const int hw_base = blockM - b_img * IMG_HW;
    const int x_img   = b_img * C_IN * IMG_HW;

    // ---- per-thread im2col geometry ----
    const int m_loc   = tid & 63;
    const int kk_base = (tid >> 6) * 8;              // {0,8,16,24}
    const int hw      = hw_base + m_loc;
    const int hrow    = hw / IMG_W;
    const int wcol    = hw - hrow * IMG_W;

    // ---- per-wave fragment geometry ----
    const int aOffBase = (wm * 32 + laneLo) * LDS_STRIDE + laneHi * 16;
    const int nBase    = wn * 64 + laneLo;           // + tb*16

    v8f acc[2][4];
#pragma unroll
    for (int ta = 0; ta < 2; ++ta)
#pragma unroll
        for (int tb = 0; tb < 4; ++tb)
#pragma unroll
            for (int i = 0; i < 8; ++i) acc[ta][tb][i] = 0.0f;

    // ---- prologue: TDM B slice 0, im2col A slice 0 ----
    float vals[8];
#if USE_TDM
    if (wave == 0) tdm_load_B(wq, 0, &lB[0][0]);
#endif
    im2col_load(x, x_img, hrow, wcol, kk_base, vals);
    im2col_store(lA[0], m_loc, kk_base, vals);
#if USE_TDM
    if (wave == 0) __builtin_amdgcn_s_wait_tensorcnt(0);
#endif
    __syncthreads();

    for (int it = 0; it < NITER; ++it) {
        const int k0       = it * KSTEP;
        const int cur      = it & 1;
        const bool hasNext = (it + 1) < NITER;

        // kick off next K-step's data movement (overlaps the WMMAs below)
        if (hasNext) {
#if USE_TDM
            if (wave == 0) tdm_load_B(wq, k0 + KSTEP, &lB[cur ^ 1][0]);
#endif
            im2col_load(x, x_img, hrow, wcol, k0 + KSTEP + kk_base, vals);
        }

        // ---- A fragments: 16-bit 16x32 layout, two 16B LDS reads each ----
        union { uint4 u[2]; v16bf v; } Af[2];
#pragma unroll
        for (int ta = 0; ta < 2; ++ta) {
            const __bf16* ab = &lA[cur][aOffBase + ta * 16 * LDS_STRIDE];
            Af[ta].u[0] = *(const uint4*)(ab);        // K = laneHi*16 .. +7
            Af[ta].u[1] = *(const uint4*)(ab + 8);    // K = laneHi*16+8 .. +15
        }

#pragma unroll
        for (int tb = 0; tb < 4; ++tb) {
            union { uint4 u[2]; v16bf v; } Bf;
#if USE_TDM
            // LDS B tile is n-major [n][32]: 32 contiguous bytes per lane
            const __bf16* bb = &lB[cur][(nBase + tb * 16) * KSTEP + laneHi * 16];
            Bf.u[0] = *(const uint4*)(bb);
            Bf.u[1] = *(const uint4*)(bb + 8);
#else
            const __bf16* bp = wq + (nBase + tb * 16) * KDIM + k0 + laneHi * 16;
            Bf.u[0] = *(const uint4*)(bp);
            Bf.u[1] = *(const uint4*)(bp + 8);
            if (tb == 0 && hasNext)
                __builtin_prefetch(bp + KSTEP, 0, 3);
#endif
#pragma unroll
            for (int ta = 0; ta < 2; ++ta)
                acc[ta][tb] = __builtin_amdgcn_wmma_f32_16x16x32_bf16(
                    false, Af[ta].v, false, Bf.v, (short)0, acc[ta][tb], false, false);
        }

        // ---- stage next A tile; drain TDM before publishing via barrier ----
        if (hasNext)
            im2col_store(lA[cur ^ 1], m_loc, kk_base, vals);
#if USE_TDM
        if (wave == 0) __builtin_amdgcn_s_wait_tensorcnt(0);
#endif
        __syncthreads();
    }

    // ---- epilogue: lane's 8 C values are consecutive in hw -> 2x float4 ----
#pragma unroll
    for (int ta = 0; ta < 2; ++ta) {
#pragma unroll
        for (int tb = 0; tb < 4; ++tb) {
            const int n = nBase + tb * 16;
            float* op = out + (b_img * C_OUT + n) * IMG_HW
                            + hw_base + wm * 32 + ta * 16 + 8 * laneHi;
            float4 lo = { acc[ta][tb][0], acc[ta][tb][1],
                          acc[ta][tb][2], acc[ta][tb][3] };
            float4 hi = { acc[ta][tb][4], acc[ta][tb][5],
                          acc[ta][tb][6], acc[ta][tb][7] };
            *(float4*)(op)     = lo;
            *(float4*)(op + 4) = hi;
        }
    }
}

// ---------------------------------------------------------------------------
// Host-side launcher
// ---------------------------------------------------------------------------
extern "C" void kernel_launch(void* const* d_in, const int* in_sizes, int n_in,
                              void* d_out, int out_size, void* d_ws, size_t ws_size,
                              hipStream_t stream)
{
    const float* x       = (const float*)d_in[0];   // (32,256,56,56)
    const float* p_c     = (const float*)d_in[1];   // (256,256,3,3,7)
    const float* q_level = (const float*)d_in[2];   // (7)
    float* out           = (float*)d_out;           // (32,256,56,56)
    __bf16* wq           = (__bf16*)d_ws;           // bf16 [256][2304] = 1.18 MB

    quantize_weights<<<2304, 256, 0, stream>>>(p_c, q_level, wq);

    conv_wmma<<<M_TOTAL / BLK_M, 256, 0, stream>>>(x, wq, out); // 1568 blocks
}
